// CrossAttention_51900384805310
// MI455X (gfx1250) — matrix operand order repaired and checked by reading, not compiled, not run on hardware
//
#include <hip/hip_runtime.h>
#include <hip/hip_bf16.h>
#include <math.h>

#define HIDDEN    1024
#define NUM_HEADS 16
#define HEAD_DIM  64
#define B_SZ      2
#define SQ        2048
#define SKV       2048
#define RMS_EPS   1e-5f
#define ATT_WAVES 4

typedef __bf16 bf16_t;
typedef __bf16 v16bf __attribute__((ext_vector_type(16)));
typedef __bf16 v8bf  __attribute__((ext_vector_type(8)));
typedef float  v8f   __attribute__((ext_vector_type(8)));
typedef int    v4i   __attribute__((ext_vector_type(4)));

#define AS1 __attribute__((address_space(1)))
#define AS3 __attribute__((address_space(3)))

// ---- CDNA5 async global->LDS DMA (ASYNCcnt-tracked), with safe fallback ----
// Probed signature (from round-2 diagnostic): parameters are v4i pointers,
// param0 in target AS1 (global): (v4i AS1*, v4i AS3*, imm offset, imm cpol).
#if defined(__HIP_DEVICE_COMPILE__) && __has_builtin(__builtin_amdgcn_global_load_async_to_lds_b128)
#define HAVE_ASYNC_LDS 1
// flat global address == AS1 address numerically; flat LDS address truncates
// to the 32-bit AS3 offset (ISA 00_overview aperture rules).
#define ASYNC_COPY_B128(gp, lp)                                              \
  __builtin_amdgcn_global_load_async_to_lds_b128(                            \
      (AS1 v4i*)(size_t)(const void*)(gp),                                   \
      (AS3 v4i*)(unsigned int)(size_t)(void*)(lp), 0, 0)
#else
#define HAVE_ASYNC_LDS 0
#endif

static __device__ __forceinline__ void async_wait_le8() {
#if HAVE_ASYNC_LDS
#if __has_builtin(__builtin_amdgcn_s_wait_asynccnt)
  __builtin_amdgcn_s_wait_asynccnt(8);
#else
  asm volatile("s_wait_asynccnt 8" ::: "memory");
#endif
#endif
}

static __device__ __forceinline__ v8f wmma_bf16(v16bf a, v16bf b, v8f c) {
  // D = A(16x32) * B(32x16) + C, f32 accumulate
  return __builtin_amdgcn_wmma_f32_16x16x32_bf16(false, a, false, b, (short)0, c,
                                                 false, false);
}

// A fragment: 16x32 (MxK) tile at `base` (row-major, leading dim `ld` elems).
static __device__ __forceinline__ v16bf load_a_frag(const bf16_t* base, int ld) {
  const int lane = threadIdx.x & 31;
  const int m  = lane & 15;
  const int kb = (lane >> 4) * 16;
  const bf16_t* p = base + (size_t)m * ld + kb;
  v8bf lo = *(const v8bf*)(p);
  v8bf hi = *(const v8bf*)(p + 8);
  v16bf f;
#pragma unroll
  for (int i = 0; i < 8; ++i) { f[i] = lo[i]; f[8 + i] = hi[i]; }
  return f;
}

// B fragment for logical B[k][n] = Wt[n][k], Wt row-major: one 32B load/lane.
static __device__ __forceinline__ v16bf load_bT_frag(const bf16_t* base, int ld) {
  const int lane = threadIdx.x & 31;
  const int n  = lane & 15;
  const int kb = (lane >> 4) * 16;
  return *(const v16bf*)(base + (size_t)n * ld + kb);
}

// B fragment for row-major B[k][n] read out of an LDS-resident 32x64 V tile.
static __device__ __forceinline__ v16bf load_bv_lds(const bf16_t* tile, int d0) {
  const int lane = threadIdx.x & 31;
  const int n  = lane & 15;
  const int kb = (lane >> 4) * 16;
  v16bf f;
#pragma unroll
  for (int i = 0; i < 16; ++i) f[i] = tile[(kb + i) * HEAD_DIM + d0 + n];
  return f;
}

// Stage one 32-key x 64-dim bf16 V tile (4KB) into LDS.
// Async path: 8 wave-wide GLOBAL_LOAD_ASYNC_TO_LDS_B128 ops (512B each).
static __device__ __forceinline__ void vtile_fetch(const bf16_t* g, bf16_t* l,
                                                   int ld2) {
  const int lane = threadIdx.x & 31;
  const int key = lane >> 3;   // 0..3
  const int seg = lane & 7;    // 16B segment within a 128B key row
  const bf16_t* gp = g + (size_t)key * ld2 + seg * 8;
  bf16_t* lp = l + key * HEAD_DIM + seg * 8;
#pragma unroll
  for (int i = 0; i < 8; ++i) {
#if HAVE_ASYNC_LDS
    ASYNC_COPY_B128(gp + (size_t)(4 * i) * ld2, lp + 4 * i * HEAD_DIM);
#else
    *(v8bf*)(lp + 4 * i * HEAD_DIM) = *(const v8bf*)(gp + (size_t)(4 * i) * ld2);
#endif
  }
}

// ---------------- prep kernels ----------------

__global__ __launch_bounds__(256)
void rmsnorm_to_bf16(const float* __restrict__ X, const float* __restrict__ w,
                     bf16_t* __restrict__ Y) {
  const int row = blockIdx.x;
  const float* x = X + (size_t)row * HIDDEN;
  float ss = 0.f;
  for (int i = threadIdx.x; i < HIDDEN; i += 256) { float v = x[i]; ss += v * v; }
#pragma unroll
  for (int m = 1; m < 32; m <<= 1) ss += __shfl_xor(ss, m, 32);
  __shared__ float red[8];
  if ((threadIdx.x & 31) == 0) red[threadIdx.x >> 5] = ss;
  __syncthreads();
  float tot = 0.f;
#pragma unroll
  for (int j = 0; j < 8; ++j) tot += red[j];
  const float inv = rsqrtf(tot * (1.0f / HIDDEN) + RMS_EPS);
  for (int i = threadIdx.x; i < HIDDEN; i += 256)
    Y[(size_t)row * HIDDEN + i] = (bf16_t)(x[i] * inv * w[i]);
}

__global__ __launch_bounds__(256)
void cvt_f32_bf16(const float* __restrict__ X, bf16_t* __restrict__ Y, size_t n) {
  size_t i = (size_t)blockIdx.x * blockDim.x + threadIdx.x;
  const size_t stride = (size_t)gridDim.x * blockDim.x;
  for (; i < n; i += stride) Y[i] = (bf16_t)X[i];
}

// ---------------- GEMM: C[M,N] = A[M,K] @ Wt[N,K]^T + bias ----------------

__global__ __launch_bounds__(256)
void gemm_bf16_wmma(const bf16_t* __restrict__ A, const bf16_t* __restrict__ Wt,
                    const float* __restrict__ bias,
                    bf16_t* __restrict__ Cb, float* __restrict__ Cf,
                    int M, int N, int K) {
  const int lane = threadIdx.x & 31;
  const int wave = threadIdx.x >> 5;
  const int wm = wave & 3, wn = wave >> 2;
  const int m0 = blockIdx.x * 128 + wm * 32;
  const int n0 = blockIdx.y * 64 + wn * 32;
  v8f acc[2][2] = {};
  for (int k0 = 0; k0 < K; k0 += 32) {
    v16bf a0 = load_a_frag(A + (size_t)m0 * K + k0, K);
    v16bf a1 = load_a_frag(A + (size_t)(m0 + 16) * K + k0, K);
    v16bf b0 = load_bT_frag(Wt + (size_t)n0 * K + k0, K);
    v16bf b1 = load_bT_frag(Wt + (size_t)(n0 + 16) * K + k0, K);
    acc[0][0] = wmma_bf16(a0, b0, acc[0][0]);
    acc[0][1] = wmma_bf16(a0, b1, acc[0][1]);
    acc[1][0] = wmma_bf16(a1, b0, acc[1][0]);
    acc[1][1] = wmma_bf16(a1, b1, acc[1][1]);
  }
  const int col = lane & 15;
  const int rb  = (lane >> 4) * 8;
#pragma unroll
  for (int i = 0; i < 2; ++i)
#pragma unroll
    for (int j = 0; j < 2; ++j) {
      const int n = n0 + j * 16 + col;
      const float bv = bias[n];
#pragma unroll
      for (int r = 0; r < 8; ++r) {
        const int m = m0 + i * 16 + rb + r;
        const float v = acc[i][j][r] + bv;
        if (Cf) Cf[(size_t)m * N + n] = v;
        else    Cb[(size_t)m * N + n] = (bf16_t)v;
      }
    }
}

// ---------------- flash attention: 1 wave = 16 queries of one (b,h) ---------
// V tiles are DMA'd into double-buffered LDS with async-to-LDS ops and the
// prefetch of tile i+1 overlaps the S-WMMA/softmax work of tile i.

__global__ __launch_bounds__(ATT_WAVES * 32)
void attn_wmma(const bf16_t* __restrict__ Qp, const bf16_t* __restrict__ KVp,
               bf16_t* __restrict__ O) {
  __shared__ __align__(32) bf16_t lds_p[ATT_WAVES * 16 * 32];
  __shared__ __align__(32) bf16_t lds_v[ATT_WAVES * 2 * 32 * HEAD_DIM];
  const int lane = threadIdx.x & 31;
  const int wave = threadIdx.x >> 5;
  const int b = blockIdx.x / NUM_HEADS;
  const int h = blockIdx.x % NUM_HEADS;
  const int qt = (blockIdx.y * ATT_WAVES + wave) * 16;
  const int ld2 = 2 * HIDDEN;
  const bf16_t* Qb = Qp + ((size_t)b * SQ + qt) * HIDDEN + h * HEAD_DIM;
  const bf16_t* Kb = KVp + (size_t)b * SKV * ld2 + h * HEAD_DIM;   // k half
  const bf16_t* Vb = Kb + HIDDEN;                                   // v half
  bf16_t* lp = lds_p + wave * (16 * 32);
  bf16_t* vbuf0 = lds_v + wave * (2 * 32 * HEAD_DIM);
  bf16_t* vbuf1 = vbuf0 + 32 * HEAD_DIM;

  const v16bf aq0 = load_a_frag(Qb, HIDDEN);       // head-dim 0..31
  const v16bf aq1 = load_a_frag(Qb + 32, HIDDEN);  // head-dim 32..63

  float rm[8], rs[8];
#pragma unroll
  for (int r = 0; r < 8; ++r) { rm[r] = -INFINITY; rs[r] = 0.f; }
  v8f acc[4] = {};

  const int col = lane & 15;
  const int rb  = (lane >> 4) * 8;

  vtile_fetch(Vb, vbuf0, ld2);               // prefetch V tile 0
  int ibuf = 0;

  for (int kv0 = 0; kv0 < SKV; kv0 += 32) {
    // prefetch next V tile into the other buffer (wraps on last iteration to
    // keep the asynccnt wait immediate and barrier trip counts uniform)
    const int nkv = (kv0 + 32) & (SKV - 1);
    vtile_fetch(Vb + (size_t)nkv * ld2, ibuf ? vbuf0 : vbuf1, ld2);

    // S = Q * K^T for 32 keys; K^T column = contiguous key row in memory
    const bf16_t* K0 = Kb + (size_t)kv0 * ld2;
    const bf16_t* K1 = K0 + (size_t)16 * ld2;
    v8f s0 = {}; v8f s1 = {};
    s0 = wmma_bf16(aq0, load_bT_frag(K0, ld2), s0);
    s0 = wmma_bf16(aq1, load_bT_frag(K0 + 32, ld2), s0);
    s1 = wmma_bf16(aq0, load_bT_frag(K1, ld2), s1);
    s1 = wmma_bf16(aq1, load_bT_frag(K1 + 32, ld2), s1);

    // online softmax; row r spans the 16 lanes of this half-wave
#pragma unroll
    for (int r = 0; r < 8; ++r) {
      const float x0 = s0[r] * 0.125f;   // 1/sqrt(64)
      const float x1 = s1[r] * 0.125f;
      float mx = fmaxf(x0, x1);
#pragma unroll
      for (int msk = 1; msk < 16; msk <<= 1)
        mx = fmaxf(mx, __shfl_xor(mx, msk, 32));
      const float mn   = fmaxf(rm[r], mx);
      const float corr = __expf(rm[r] - mn);
      rm[r] = mn;
      const float e0 = __expf(x0 - mn);
      const float e1 = __expf(x1 - mn);
      float ps = e0 + e1;
#pragma unroll
      for (int msk = 1; msk < 16; msk <<= 1) ps += __shfl_xor(ps, msk, 32);
      rs[r] = rs[r] * corr + ps;
#pragma unroll
      for (int d = 0; d < 4; ++d) acc[d][r] *= corr;
      s0[r] = e0; s1[r] = e1;
    }

    // C-layout P -> row-major LDS patch -> A-layout fragment
#pragma unroll
    for (int r = 0; r < 8; ++r) {
      lp[(rb + r) * 32 + col]      = (bf16_t)s0[r];
      lp[(rb + r) * 32 + 16 + col] = (bf16_t)s1[r];
    }
    __syncthreads();
    const v16bf ap = load_a_frag(lp, 32);
    __syncthreads();

    // current V tile ready once <=8 async ops (the next tile's) remain
    async_wait_le8();
    const bf16_t* vt = ibuf ? vbuf1 : vbuf0;
#pragma unroll
    for (int d = 0; d < 4; ++d)
      acc[d] = wmma_bf16(ap, load_bv_lds(vt, d * 16), acc[d]);
    ibuf ^= 1;
  }

  bf16_t* Ob = O + ((size_t)b * SQ + qt) * HIDDEN + h * HEAD_DIM;
#pragma unroll
  for (int r = 0; r < 8; ++r) {
    const float inv = 1.f / rs[r];
#pragma unroll
    for (int d = 0; d < 4; ++d)
      Ob[(size_t)(rb + r) * HIDDEN + d * 16 + col] = (bf16_t)(acc[d][r] * inv);
  }
}

// ---------------- launch ----------------

extern "C" void kernel_launch(void* const* d_in, const int* in_sizes, int n_in,
                              void* d_out, int out_size, void* d_ws, size_t ws_size,
                              hipStream_t stream) {
  const float* q      = (const float*)d_in[0];
  const float* kv     = (const float*)d_in[1];
  const float* w_norm = (const float*)d_in[2];
  const float* w_q    = (const float*)d_in[3];
  const float* b_q    = (const float*)d_in[4];
  const float* w_kv   = (const float*)d_in[5];
  const float* b_kv   = (const float*)d_in[6];
  const float* w_out  = (const float*)d_in[7];
  const float* b_out  = (const float*)d_in[8];
  float* out = (float*)d_out;

  char* ws = (char*)d_ws;
  const size_t MB = 1024u * 1024u;
  const size_t M  = (size_t)B_SZ * SQ;                 // 4096 rows
  bf16_t* qn   = (bf16_t*)(ws);                        // [4096,1024]   8 MB
  bf16_t* kvb  = (bf16_t*)(ws + 8 * MB);               // [4096,1024]   8 MB
  bf16_t* wqb  = (bf16_t*)(ws + 16 * MB);              // [1024,1024]   2 MB
  bf16_t* wkvb = (bf16_t*)(ws + 18 * MB);              // [2048,1024]   4 MB
  bf16_t* wob  = (bf16_t*)(ws + 22 * MB);              // [1024,1024]   2 MB
  bf16_t* qp   = (bf16_t*)(ws + 24 * MB);              // [4096,1024]   8 MB
  bf16_t* kvp  = (bf16_t*)(ws + 32 * MB);              // [4096,2048]  16 MB
  bf16_t* ob   = (bf16_t*)(ws + 48 * MB);              // [4096,1024]   8 MB

  rmsnorm_to_bf16<<<(int)M, 256, 0, stream>>>(q, w_norm, qn);
  cvt_f32_bf16<<<2048, 256, 0, stream>>>(kv, kvb, M * HIDDEN);
  cvt_f32_bf16<<<1024, 256, 0, stream>>>(w_q, wqb, (size_t)HIDDEN * HIDDEN);
  cvt_f32_bf16<<<1024, 256, 0, stream>>>(w_kv, wkvb, (size_t)2 * HIDDEN * HIDDEN);
  cvt_f32_bf16<<<1024, 256, 0, stream>>>(w_out, wob, (size_t)HIDDEN * HIDDEN);

  dim3 blk(256);
  dim3 g_q((unsigned)(M / 128), HIDDEN / 64);
  gemm_bf16_wmma<<<g_q, blk, 0, stream>>>(qn, wqb, b_q, qp, nullptr,
                                          (int)M, HIDDEN, HIDDEN);
  dim3 g_kv((unsigned)(M / 128), (2 * HIDDEN) / 64);
  gemm_bf16_wmma<<<g_kv, blk, 0, stream>>>(kvb, wkvb, b_kv, kvp, nullptr,
                                           (int)M, 2 * HIDDEN, HIDDEN);

  dim3 g_att(B_SZ * NUM_HEADS, SQ / 16 / ATT_WAVES);
  attn_wmma<<<g_att, dim3(ATT_WAVES * 32), 0, stream>>>(qp, kvp, ob);

  gemm_bf16_wmma<<<g_q, blk, 0, stream>>>(ob, wob, b_out, nullptr, out,
                                          (int)M, HIDDEN, HIDDEN);
}
